// LSTM_12738873000535
// MI455X (gfx1250) — matrix operand order repaired and checked by reading, not compiled, not run on hardware
//
#include <hip/hip_runtime.h>

typedef __attribute__((ext_vector_type(16))) _Float16 v16h;
typedef __attribute__((ext_vector_type(8)))  float    v8f;

#define B_SZ 8192
#define T_SZ 512
#define D_IN 8
#define HID  32
#define TCHUNK 16

// sigmoid(x) = 1 / (1 + 2^(-x*log2e)) ; tanh(x) = 1 - 2/(1 + 2^(2x*log2e))
__device__ __forceinline__ float fast_sigmoid(float x) {
    return __builtin_amdgcn_rcpf(1.0f + __builtin_amdgcn_exp2f(-1.4426950408889634f * x));
}
__device__ __forceinline__ float fast_tanh(float x) {
    return 1.0f - 2.0f * __builtin_amdgcn_rcpf(1.0f + __builtin_amdgcn_exp2f(2.8853900817779268f * x));
}

// Block = 64 threads = 2 waves. Wave w handles batch rows [blockIdx.x*16, +16)
// and 16 gate columns (w*16 .. w*16+15) of each gate i/f/g/o.
__global__ __launch_bounds__(64) void lstm_fused(
    const float* __restrict__ x,    // [B, T, D_IN]
    const float* __restrict__ W_ih, // [4H, D_IN]
    const float* __restrict__ W_hh, // [4H, H]
    const float* __restrict__ b_ih, // [4H]
    const float* __restrict__ b_hh, // [4H]
    const float* __restrict__ W_fc, // [1, H]
    const float* __restrict__ b_fc, // [1]
    float* __restrict__ out)        // [B]  (== [1,B,1] flat)
{
    __shared__ __align__(16) _Float16 hbuf[2][16 * HID];          // h tile [m][k], double buffered
    __shared__ __align__(16) _Float16 xstage[TCHUNK][16][D_IN];   // x chunk, f16, [t_loc][m][k]
    __shared__ float out_acc[16];

    const int tid     = threadIdx.x;
    const int w       = tid >> 5;   // wave id within block (0/1)
    const int lane    = tid & 31;
    const int lane_lo = lane & 15;
    const int grp     = lane >> 4;  // lane half-group
    const int base    = blockIdx.x * 16;

    if (tid < 16) out_acc[tid] = 0.0f;

    // ---- One-time: gather weights into WMMA B-operand registers (f16 32x16 tiles).
    // B layout: N column = lane%16, element i = K (table order: 8*grp + i, +8 for i>=8).
    v16h bhh[4];  // h path:  W_hh^T tile for gate tt, cols ncol0..ncol0+15, K=0..31
    v16h bx[4];   // x path:  W_ih^T zero-padded from K=8 to K=32
    float bias[4];
    #pragma unroll
    for (int tt = 0; tt < 4; ++tt) {
        const int ncol0 = tt * HID + w * 16;
        #pragma unroll
        for (int i = 0; i < 16; ++i) {
            const int kk = grp * 8 + i + ((i >= 8) ? 8 : 0);
            bhh[tt][i] = (_Float16)W_hh[(ncol0 + lane_lo) * HID + kk];
            bx[tt][i]  = (kk < D_IN) ? (_Float16)W_ih[(ncol0 + lane_lo) * D_IN + kk]
                                     : (_Float16)0.0f;
        }
        bias[tt] = b_ih[ncol0 + lane_lo] + b_hh[ncol0 + lane_lo];
    }

    // ---- Persistent state in registers.
    v8f c_st, h_keep;            // cell state / last h, C-layout (m = e + 8*grp, n = lane_lo)
    #pragma unroll
    for (int e = 0; e < 8; ++e) { c_st[e] = 0.0f; h_keep[e] = 0.0f; }
    v16h hA;                     // h as WMMA A-operand (16x32 f16), starts at 0
    #pragma unroll
    for (int i = 0; i < 16; ++i) hA[i] = (_Float16)0.0f;

    const int hk   = w * 16 + lane_lo;  // hidden column this lane produces
    const int xrow = tid >> 2;          // staging: row handled by this thread
    const int xseg = tid & 3;           // staging: which 4-timestep segment

    __syncthreads();

    for (int tc = 0; tc < T_SZ; tc += TCHUNK) {
        // ---- Stage x chunk: rows are read fully-contiguously (512B/row) and
        // converted to f16 once; LDS layout [t_loc][m][k] matches per-step fetch.
        {
            const float4* px =
                (const float4*)(x + ((size_t)(base + xrow) * T_SZ + tc + xseg * 4) * D_IN);
            #pragma unroll
            for (int j = 0; j < 4; ++j) {
                float4 f0 = px[2 * j], f1 = px[2 * j + 1];
                union { _Float16 h[8]; uint4 q; } u;
                u.h[0] = (_Float16)f0.x; u.h[1] = (_Float16)f0.y;
                u.h[2] = (_Float16)f0.z; u.h[3] = (_Float16)f0.w;
                u.h[4] = (_Float16)f1.x; u.h[5] = (_Float16)f1.y;
                u.h[6] = (_Float16)f1.z; u.h[7] = (_Float16)f1.w;
                *(uint4*)&xstage[xseg * 4 + j][xrow][0] = u.q;
            }
        }
        __syncthreads();

        for (int ti = 0; ti < TCHUNK; ++ti) {
            const int t = tc + ti;

            // A_x operand: 16x32 f16, x_t occupies K=0..7 (lanes 0..15), rest zero.
            v16h ax;
            #pragma unroll
            for (int i = 0; i < 16; ++i) ax[i] = (_Float16)0.0f;
            if (lane < 16) {
                union { _Float16 h[8]; uint4 q; } u;
                u.q = *(const uint4*)&xstage[ti][lane][0];   // ds_load_b128
                #pragma unroll
                for (int i = 0; i < 8; ++i) ax[i] = u.h[i];
            }

            // gates = bias + x_t @ W_ih^T + h @ W_hh^T   (two K=32 WMMAs per gate tile)
            v8f acc[4];
            #pragma unroll
            for (int tt = 0; tt < 4; ++tt) {
                v8f a;
                #pragma unroll
                for (int e = 0; e < 8; ++e) a[e] = bias[tt];
                a = __builtin_amdgcn_wmma_f32_16x16x32_f16(false, ax, false, bx[tt],
                                                           (short)0, a, false, false);
                a = __builtin_amdgcn_wmma_f32_16x16x32_f16(false, hA, false, bhh[tt],
                                                           (short)0, a, false, false);
                acc[tt] = a;
            }

            // Elementwise LSTM cell update; tiles 0..3 = i,f,g,o share C layout with c_st.
            _Float16* hrow = &hbuf[t & 1][0];
            #pragma unroll
            for (int e = 0; e < 8; ++e) {
                float iv = fast_sigmoid(acc[0][e]);
                float fv = fast_sigmoid(acc[1][e]);
                float gv = fast_tanh   (acc[2][e]);
                float ov = fast_sigmoid(acc[3][e]);
                float cv = fv * c_st[e] + iv * gv;
                c_st[e]  = cv;
                float hv = ov * fast_tanh(cv);
                h_keep[e] = hv;
                hrow[(e + 8 * grp) * HID + hk] = (_Float16)hv; // ds_store_b16
            }
            __syncthreads();

            // Repack h (row-major 16x32 f16 in LDS) into A-operand layout for next step:
            // lanes 0-15: K 0-7 / 16-23 ; lanes 16-31: K 8-15 / 24-31.
            const _Float16* hb = &hbuf[t & 1][0];
            union { v16h v; uint4 q[2]; } u;
            u.q[0] = *(const uint4*)(hb + lane_lo * HID + grp * 8);       // halves 0..7
            u.q[1] = *(const uint4*)(hb + lane_lo * HID + grp * 8 + 16);  // halves 8..15
            hA = u.v;
        }
    }

    // ---- out[m] = sum_n h[m][n] * W_fc[n] + b_fc ; reduce 16 lanes + 2 waves via LDS.
    const float wfc = W_fc[hk];
    v8f part;
    #pragma unroll
    for (int e = 0; e < 8; ++e) part[e] = h_keep[e] * wfc;
    #pragma unroll
    for (int mask = 1; mask <= 8; mask <<= 1) {
        #pragma unroll
        for (int e = 0; e < 8; ++e) part[e] += __shfl_xor(part[e], mask, 32);
    }
    if (lane_lo == 0) {
        #pragma unroll
        for (int e = 0; e < 8; ++e) atomicAdd(&out_acc[e + 8 * grp], part[e]);
    }
    __syncthreads();
    if (tid < 16) out[base + tid] = out_acc[tid] + b_fc[0];
}

extern "C" void kernel_launch(void* const* d_in, const int* in_sizes, int n_in,
                              void* d_out, int out_size, void* d_ws, size_t ws_size,
                              hipStream_t stream) {
    (void)in_sizes; (void)n_in; (void)d_ws; (void)ws_size; (void)out_size;
    const float* x    = (const float*)d_in[0];
    const float* W_ih = (const float*)d_in[1];
    const float* W_hh = (const float*)d_in[2];
    const float* b_ih = (const float*)d_in[3];
    const float* b_hh = (const float*)d_in[4];
    const float* W_fc = (const float*)d_in[5];
    const float* b_fc = (const float*)d_in[6];
    float* out = (float*)d_out;

    dim3 grid(B_SZ / 16);  // 512 blocks
    dim3 block(64);        // 2 waves, wave32
    lstm_fused<<<grid, block, 0, stream>>>(x, W_ih, W_hh, b_ih, b_hh, W_fc, b_fc, out);
}
